// DecoderRNN_60739427500653
// MI455X (gfx1250) — compile-verified, matrix operand order checked by reading, not run on hardware
//
#include <hip/hip_runtime.h>

// ---------------------------------------------------------------------------
// LSTM decoder (DecoderRNN) for MI455X / gfx1250, wave32 + WMMA bf16.
//
//  * fc GEMM (128x1024x32000 per step, 31 steps) dominates FLOPs (~260 GFLOP).
//  * Output tensor (524 MB f32) sets the HBM floor (~22.5us @ 23.3 TB/s).
//  * fc_W converted once to bf16 (64 MB) -> resident in 192 MB L2 across all
//    31 steps. Output written with non-temporal stores to protect L2.
//  * bf16 v_wmma_f32_16x16x32_bf16 (f32 accumulate): 8x denser per instruction
//    than f32 16x16x4 WMMA.
//  * fc kernel: 4 vocab tiles per wave sharing one A fragment; the block's A
//    tile (16 x 1024 h-rows) is staged into LDS with the gfx1250 async
//    global->LDS path (ASYNCcnt) and read back via ds_load_b128 with a padded
//    stride (bank-conflict-free fragment reads).
// ---------------------------------------------------------------------------

typedef __attribute__((ext_vector_type(16))) __bf16 bf16x16;
typedef __attribute__((ext_vector_type(8)))  __bf16 bf16x8;
typedef __attribute__((ext_vector_type(8)))  float  f32x8;

#define B_ 128
#define T_ 32
#define E_ 512
#define H_ 1024
#define V_ 32000
#define SA_STRIDE 1032   // 1024 bf16 + 8 pad -> 2064B row stride, bank-friendly

__device__ __forceinline__ unsigned short f2bf(float f) {
  unsigned int u = __float_as_uint(f);
  u += 0x7FFFu + ((u >> 16) & 1u);          // round to nearest even
  return (unsigned short)(u >> 16);
}

// Lane fragment: two contiguous 16B chunks, elements 0..7 at p[0..7],
// elements 8..15 at p[16..23] (A/B 16-bit 16x32 WMMA layout).
__device__ __forceinline__ bf16x16 load_frag(const unsigned short* p) {
  bf16x16 v;
  *(bf16x8*)&v         = *(const bf16x8*)p;
  *(((bf16x8*)&v) + 1) = *(const bf16x8*)(p + 16);
  return v;
}

__device__ __forceinline__ float sigm(float x) { return 1.0f / (1.0f + __expf(-x)); }

// ---------------------------------------------------------------- prep kernels

__global__ void k_f32_to_bf16(const float* __restrict__ src,
                              unsigned short* __restrict__ dst, int n) {
  int i = blockIdx.x * blockDim.x + threadIdx.x;
  int stride = gridDim.x * blockDim.x;
  for (; i < n; i += stride) dst[i] = f2bf(src[i]);
}

// xbf[t][b][e] = bf16(embed_table[captions[b][t]][e]), t in [0, T-2]
__global__ void k_embed(const int* __restrict__ captions,
                        const float* __restrict__ table,
                        unsigned short* __restrict__ xbf) {
  long i = (long)blockIdx.x * blockDim.x + threadIdx.x;
  const long n = (long)(T_ - 1) * B_ * E_;
  if (i >= n) return;
  int e = (int)(i % E_);
  int b = (int)((i / E_) % B_);
  int t = (int)(i / ((long)E_ * B_));
  int idx = captions[b * T_ + t];
  xbf[i] = f2bf(table[(long)idx * E_ + e]);
}

__global__ void k_init_state(const float* __restrict__ feat,
                             float* __restrict__ cst,
                             unsigned short* __restrict__ hbf) {
  int i = blockIdx.x * blockDim.x + threadIdx.x;
  if (i >= B_ * H_) return;
  float v = feat[i];
  cst[i] = v;
  hbf[i] = f2bf(v);
}

__global__ void k_zero_t0(float* __restrict__ out) {
  long i = (long)blockIdx.x * blockDim.x + threadIdx.x;
  if (i >= (long)B_ * V_) return;
  long b = i / V_;
  long v = i % V_;
  __builtin_nontemporal_store(0.0f, &out[(b * T_) * V_ + v]);
}

// ------------------------------------------------------------- LSTM step GEMM
// gates[128, 4*1024] = x_t @ W_ih^T + h @ W_hh^T + b, fused with activations
// and (h, c) state update. One wave owns one 16x16 hidden tile for ALL FOUR
// gates (they share the A fragment), so the elementwise LSTM update is local.
__global__ void __launch_bounds__(128)
k_lstm_step(const unsigned short* __restrict__ xbf_t,   // [B_][E_]
            const unsigned short* __restrict__ wih,     // [4H][E_] bf16
            const unsigned short* __restrict__ whh,     // [4H][H_] bf16
            const float* __restrict__ b_ih,
            const float* __restrict__ b_hh,
            const unsigned short* __restrict__ hbf_in,  // [B_][H_] bf16
            float* __restrict__ cst,                    // [B_][H_] f32 (in-place)
            unsigned short* __restrict__ hbf_out)       // [B_][H_] bf16
{
  const int lane = threadIdx.x & 31;
  const int wid  = threadIdx.x >> 5;
  const int m0   = blockIdx.x * 16;                 // batch tile
  const int n0   = (blockIdx.y * 4 + wid) * 16;     // hidden tile (per gate)
  const int half = lane >> 4;
  const int l15  = lane & 15;

  f32x8 acc[4] = {};                                // i, f, g, o accumulators

  // x @ W_ih^T contribution (K = 512)
  const unsigned short* arow = xbf_t + (size_t)(m0 + l15) * E_ + half * 8;
  #pragma unroll 2
  for (int k = 0; k < E_; k += 32) {
    bf16x16 a = load_frag(arow + k);
    #pragma unroll
    for (int g = 0; g < 4; ++g) {
      const unsigned short* bp =
          wih + (size_t)(g * H_ + n0 + l15) * E_ + k + half * 8;
      bf16x16 b = load_frag(bp);
      acc[g] = __builtin_amdgcn_wmma_f32_16x16x32_bf16(
          false, a, false, b, (short)0, acc[g], false, false);
    }
  }

  // h @ W_hh^T contribution (K = 1024)
  const unsigned short* hrow = hbf_in + (size_t)(m0 + l15) * H_ + half * 8;
  #pragma unroll 2
  for (int k = 0; k < H_; k += 32) {
    bf16x16 a = load_frag(hrow + k);
    #pragma unroll
    for (int g = 0; g < 4; ++g) {
      const unsigned short* bp =
          whh + (size_t)(g * H_ + n0 + l15) * H_ + k + half * 8;
      bf16x16 b = load_frag(bp);
      acc[g] = __builtin_amdgcn_wmma_f32_16x16x32_bf16(
          false, a, false, b, (short)0, acc[g], false, false);
    }
  }

  // Fused bias + activations + cell/hidden update.
  // C/D layout: col = n0 + (lane&15); row = m0 + v + 8*(lane>=16).
  const int ncol = n0 + l15;
  const float bi = b_ih[ncol]          + b_hh[ncol];
  const float bf = b_ih[H_ + ncol]     + b_hh[H_ + ncol];
  const float bg = b_ih[2 * H_ + ncol] + b_hh[2 * H_ + ncol];
  const float bo = b_ih[3 * H_ + ncol] + b_hh[3 * H_ + ncol];
  #pragma unroll
  for (int v = 0; v < 8; ++v) {
    const int m = m0 + v + half * 8;
    const size_t sidx = (size_t)m * H_ + ncol;
    float ig = sigm(acc[0][v] + bi);
    float fg = sigm(acc[1][v] + bf);
    float gg = tanhf(acc[2][v] + bg);
    float og = sigm(acc[3][v] + bo);
    float cn = fg * cst[sidx] + ig * gg;
    cst[sidx] = cn;
    hbf_out[sidx] = f2bf(og * tanhf(cn));
  }
}

// ----------------------------------------------------------------- fc (logits)
// logits[128, 32000] = h @ fc_W^T + fc_b, streamed to d_out[:, t, :].
// Block = 4 waves; block stages its 16 h-rows (32KB bf16) into LDS once via
// the async global->LDS path, then each wave computes 4 vocab tiles per
// K-step off one shared A fragment (ds_load) and 4 B fragments (global).
__global__ void __launch_bounds__(128)
k_fc(const unsigned short* __restrict__ hbf,   // [B_][H_] bf16
     const unsigned short* __restrict__ fw,    // [V_][H_] bf16 (L2-resident)
     const float* __restrict__ fb,
     float* __restrict__ out, int t_out)
{
  __shared__ unsigned short sA[16 * SA_STRIDE];

  const int tid  = threadIdx.x;
  const int lane = tid & 31;
  const int wid  = tid >> 5;
  const int m0   = blockIdx.x * 16;
  const int nt0  = blockIdx.y * 16 + wid * 4;     // first of 4 vocab tiles
  const int half = lane >> 4;
  const int l15  = lane & 15;

  // ---- stage A tile (16 x 1024 bf16) into LDS: async copy, ASYNCcnt-tracked.
  // 2048 16B-chunks total; 128 threads x 16 iterations.
  #pragma unroll
  for (int c = tid; c < 16 * 128; c += 128) {
    const int row = c >> 7;
    const int cc  = c & 127;
    unsigned dst_lds = (unsigned)(unsigned long long)&sA[row * SA_STRIDE + cc * 8];
    unsigned long long src =
        (unsigned long long)(hbf + (size_t)(m0 + row) * H_ + cc * 8);
    asm volatile("global_load_async_to_lds_b128 %0, %1, off"
                 :: "v"(dst_lds), "v"(src) : "memory");
  }
  asm volatile("s_wait_asynccnt 0" ::: "memory");
  __syncthreads();

  f32x8 acc[4] = {};
  const unsigned short* aptr = &sA[l15 * SA_STRIDE + half * 8];
  const unsigned short* br0 = fw + (size_t)((nt0 + 0) * 16 + l15) * H_ + half * 8;
  const unsigned short* br1 = fw + (size_t)((nt0 + 1) * 16 + l15) * H_ + half * 8;
  const unsigned short* br2 = fw + (size_t)((nt0 + 2) * 16 + l15) * H_ + half * 8;
  const unsigned short* br3 = fw + (size_t)((nt0 + 3) * 16 + l15) * H_ + half * 8;

  #pragma unroll 2
  for (int k = 0; k < H_; k += 32) {
    bf16x16 a = load_frag(aptr + k);   // ds_load_b128 x2, conflict-free
    bf16x16 b0 = load_frag(br0 + k);
    bf16x16 b1 = load_frag(br1 + k);
    bf16x16 b2 = load_frag(br2 + k);
    bf16x16 b3 = load_frag(br3 + k);
    acc[0] = __builtin_amdgcn_wmma_f32_16x16x32_bf16(
        false, a, false, b0, (short)0, acc[0], false, false);
    acc[1] = __builtin_amdgcn_wmma_f32_16x16x32_bf16(
        false, a, false, b1, (short)0, acc[1], false, false);
    acc[2] = __builtin_amdgcn_wmma_f32_16x16x32_bf16(
        false, a, false, b2, (short)0, acc[2], false, false);
    acc[3] = __builtin_amdgcn_wmma_f32_16x16x32_bf16(
        false, a, false, b3, (short)0, acc[3], false, false);
  }

  #pragma unroll
  for (int j = 0; j < 4; ++j) {
    const int ncol = (nt0 + j) * 16 + l15;
    const float bias = fb[ncol];
    #pragma unroll
    for (int v = 0; v < 8; ++v) {
      const int m = m0 + v + half * 8;
      // NT store: keep fc_W (64MB bf16) resident in the 192MB L2 across steps.
      __builtin_nontemporal_store(acc[j][v] + bias,
                                  &out[((size_t)m * T_ + t_out) * V_ + ncol]);
    }
  }
}

// ------------------------------------------------------------------- launcher

extern "C" void kernel_launch(void* const* d_in, const int* in_sizes, int n_in,
                              void* d_out, int out_size, void* d_ws, size_t ws_size,
                              hipStream_t stream) {
  const float* features = (const float*)d_in[0];
  const int*   captions = (const int*)d_in[1];
  const float* table    = (const float*)d_in[2];
  const float* W_ih     = (const float*)d_in[3];
  const float* W_hh     = (const float*)d_in[4];
  const float* b_ih     = (const float*)d_in[5];
  const float* b_hh     = (const float*)d_in[6];
  const float* fc_W     = (const float*)d_in[7];
  const float* fc_b     = (const float*)d_in[8];
  float* out = (float*)d_out;
  (void)in_sizes; (void)n_in; (void)out_size; (void)ws_size;

  // Workspace carve-up (~83 MB total, 256B aligned).
  char* ws = (char*)d_ws;
  size_t off = 0;
  auto carve = [&](size_t bytes) -> char* {
    char* p = ws + off;
    off = (off + bytes + 255) & ~(size_t)255;
    return p;
  };
  unsigned short* wih_bf = (unsigned short*)carve((size_t)4 * H_ * E_ * 2);
  unsigned short* whh_bf = (unsigned short*)carve((size_t)4 * H_ * H_ * 2);
  unsigned short* fcw_bf = (unsigned short*)carve((size_t)V_ * H_ * 2);
  unsigned short* xbf    = (unsigned short*)carve((size_t)(T_ - 1) * B_ * E_ * 2);
  float*          c_f32  = (float*)carve((size_t)B_ * H_ * 4);
  unsigned short* hbuf0  = (unsigned short*)carve((size_t)B_ * H_ * 2);
  unsigned short* hbuf1  = (unsigned short*)carve((size_t)B_ * H_ * 2);
  unsigned short* hbuf[2] = {hbuf0, hbuf1};

  // One-time conversions / gathers (grid-stride where large).
  k_f32_to_bf16<<<2048, 256, 0, stream>>>(W_ih, wih_bf, 4 * H_ * E_);
  k_f32_to_bf16<<<2048, 256, 0, stream>>>(W_hh, whh_bf, 4 * H_ * H_);
  k_f32_to_bf16<<<4096, 256, 0, stream>>>(fc_W, fcw_bf, V_ * H_);
  {
    long n = (long)(T_ - 1) * B_ * E_;
    k_embed<<<(unsigned)((n + 255) / 256), 256, 0, stream>>>(captions, table, xbf);
  }
  k_init_state<<<(B_ * H_ + 255) / 256, 256, 0, stream>>>(features, c_f32, hbuf[0]);
  {
    long n = (long)B_ * V_;
    k_zero_t0<<<(unsigned)((n + 255) / 256), 256, 0, stream>>>(out);
  }

  // Sequential scan over timesteps (h ping-pongs between bf16 buffers).
  for (int t = 0; t < T_ - 1; ++t) {
    const unsigned short* hin  = hbuf[t & 1];
    unsigned short*       hout = hbuf[(t + 1) & 1];
    k_lstm_step<<<dim3(B_ / 16, H_ / (16 * 4)), 128, 0, stream>>>(
        xbf + (size_t)t * B_ * E_, wih_bf, whh_bf, b_ih, b_hh,
        hin, c_f32, hout);
    // 2000 vocab tiles = grid.y(125) * 4 waves * 4 tiles
    k_fc<<<dim3(B_ / 16, V_ / (16 * 16)), 128, 0, stream>>>(
        hout, fcw_bf, fc_b, out, t + 1);
  }
}
